// IUP_BLOCK_15616501088693
// MI455X (gfx1250) — matrix operand order repaired and checked by reading, not compiled, not run on hardware
//
#include <hip/hip_runtime.h>
#include <hip/hip_bf16.h>
#include <stdint.h>

typedef __bf16 bf16;
typedef __attribute__((ext_vector_type(16))) bf16     v16bf;
typedef __attribute__((ext_vector_type(8)))  float    v8f;
typedef __attribute__((ext_vector_type(4)))  uint32_t v4u;
typedef __attribute__((ext_vector_type(8)))  int      v8i;
typedef __attribute__((ext_vector_type(4)))  int      v4i;

#define WMMA_BF16(A, B, C) \
  __builtin_amdgcn_wmma_f32_16x16x32_bf16(false, (A), false, (B), (short)0, (C), false, false)

// Problem constants
constexpr int Bn  = 8;
constexpr int CIN = 512;
constexpr int Ww  = 64;
constexpr int P   = 4096;   // 64*64 pixels per batch
constexpr int DX  = 256;    // DIM_X
constexpr int DY  = 128;    // DIM_Y
constexpr int E   = 64;     // EMB
constexpr int HU  = 128;    // 2H
constexpr int WU  = 128;    // 2W

// ---------------- WMMA fragment loaders (CDNA5 wave32 layouts) ----------------
// A 16x32 bf16: lane(m=l&15, hi=l>>4); VGPR v -> k = hi*16 + 2v{,+1}
__device__ __forceinline__ v16bf load_a_f32(const float* __restrict__ base, int ld, int k0) {
  const int lane = threadIdx.x & 31;
  const int m = lane & 15, hi = lane >> 4;
  const float* row = base + (size_t)m * ld + k0;
  v16bf a;
#pragma unroll
  for (int v = 0; v < 8; ++v) {
    const int kk = hi * 16 + v * 2;
    a[2 * v]     = (bf16)row[kk];
    a[2 * v + 1] = (bf16)row[kk + 1];
  }
  return a;
}

__device__ __forceinline__ v16bf load_a_bf(const bf16* __restrict__ base, int ld, int k0) {
  const int lane = threadIdx.x & 31;
  const int m = lane & 15, hi = lane >> 4;
  const bf16* row = base + (size_t)m * ld + k0;
  v16bf a;
#pragma unroll
  for (int v = 0; v < 8; ++v) {
    const int kk = hi * 16 + v * 2;
    a[2 * v]     = row[kk];
    a[2 * v + 1] = row[kk + 1];
  }
  return a;
}

// B 32x16 bf16: lane(n=l&15, hi=l>>4); element j -> k = hi*16 + j
__device__ __forceinline__ v16bf load_b_kmajor_f32(const float* src, int ld, int k0, int n0) {
  const int lane = threadIdx.x & 31;
  const int n = lane & 15, hi = lane >> 4;
  v16bf b;
#pragma unroll
  for (int j = 0; j < 16; ++j)
    b[j] = (bf16)src[(size_t)(k0 + hi * 16 + j) * ld + n0 + n];
  return b;
}

__device__ __forceinline__ v16bf load_b_nmajor_f32(const float* src, int ld, int k0, int n0) {
  const int lane = threadIdx.x & 31;
  const int n = lane & 15, hi = lane >> 4;
  const float* row = src + (size_t)(n0 + n) * ld + k0 + hi * 16;
  v16bf b;
#pragma unroll
  for (int j = 0; j < 16; ++j) b[j] = (bf16)row[j];
  return b;
}

__device__ __forceinline__ v16bf load_b_nmajor_bf(const bf16* src, int ld, int k0, int n0) {
  const int lane = threadIdx.x & 31;
  const int n = lane & 15, hi = lane >> 4;
  const bf16* row = src + (size_t)(n0 + n) * ld + k0 + hi * 16;
  v16bf b;
#pragma unroll
  for (int j = 0; j < 16; ++j) b[j] = row[j];
  return b;
}

// ---------------- Kernel 1: 1x1 conv  Vc[b][o][p] = Wc @ x  (+bc) ----------------
__global__ __launch_bounds__(256) void conv1x1_kernel(const float* __restrict__ x,
                                                      const float* __restrict__ Wc,
                                                      const float* __restrict__ bc,
                                                      float* __restrict__ Vc,
                                                      bf16* __restrict__ Vcb) {
  __shared__ float xs[32 * 64];  // k-slab tile: 32 channels x 64 pixels
  const int b    = blockIdx.y;
  const int p0   = blockIdx.x * 64;
  const int tid  = threadIdx.x;
  const int wave = tid >> 5;
  const int o0   = wave * 32;  // 2 M-tiles per wave
  const float* X = x + (size_t)b * CIN * P;

  v8f acc[2][4] = {};

  for (int kt = 0; kt < 16; ++kt) {
    const int k0 = kt * 32;
    __syncthreads();
#if __has_builtin(__builtin_amdgcn_tensor_load_to_lds)
    if (tid < 32) {
      // TDM: 2D tile [32 rows x 64 px] of fp32 from global row-major [512,4096] -> LDS
      const uint64_t ga = (uint64_t)(uintptr_t)(X + (size_t)k0 * P + p0);
      const uint32_t lo = (uint32_t)(uintptr_t)xs;  // LDS byte offset = addr[31:0]
      v4u g0;
      g0[0] = 1u;                                             // count = 1 valid descriptor
      g0[1] = lo;                                             // lds_addr
      g0[2] = (uint32_t)ga;                                   // global_addr[31:0]
      g0[3] = (uint32_t)((ga >> 32) & 0x1FFFFFFu) | (2u << 30);  // addr[56:32] | type=2
      v8i g1;
      g1[0] = (int)(2u << 16);                                // data_size=2 (4B)
      g1[1] = (int)(((uint32_t)P & 0xFFFFu) << 16);           // tensor_dim0 lo16
      g1[2] = (int)(((uint32_t)P >> 16) | ((uint32_t)CIN << 16));  // dim0 hi | tensor_dim1 lo
      g1[3] = (int)(64u << 16);                               // dim1 hi=0 | tile_dim0=64
      g1[4] = (int)32u;                                       // tile_dim1=32 | tile_dim2=0
      g1[5] = (int)P;                                         // tensor_dim0_stride lo32
      g1[6] = 0;
      g1[7] = 0;
      v4i gz4 = {};
      v8i gz8 = {};
      __builtin_amdgcn_tensor_load_to_lds(g0, g1, gz4, gz4, gz8, 0);
      __builtin_amdgcn_s_wait_tensorcnt(0);
    }
#else
    for (int idx = tid; idx < 32 * 64; idx += 256)
      xs[idx] = X[(size_t)(k0 + (idx >> 6)) * P + p0 + (idx & 63)];
#endif
    __syncthreads();

    v16bf a0 = load_a_f32(Wc + (size_t)o0 * CIN, CIN, k0);
    v16bf a1 = load_a_f32(Wc + (size_t)(o0 + 16) * CIN, CIN, k0);
#pragma unroll
    for (int nt = 0; nt < 4; ++nt) {
      v16bf bfrag = load_b_kmajor_f32(xs, 64, 0, nt * 16);
      acc[0][nt]  = WMMA_BF16(a0, bfrag, acc[0][nt]);
      acc[1][nt]  = WMMA_BF16(a1, bfrag, acc[1][nt]);
    }
  }

  const int lane = tid & 31, n = lane & 15, hi = lane >> 4;
#pragma unroll
  for (int i = 0; i < 2; ++i)
#pragma unroll
    for (int nt = 0; nt < 4; ++nt)
#pragma unroll
      for (int r = 0; r < 8; ++r) {
        const int o = o0 + i * 16 + r + 8 * hi;
        const int p = p0 + nt * 16 + n;
        const float v = acc[i][nt][r] + bc[o];
        const size_t off = ((size_t)b * DX + o) * P + p;
        Vc[off]  = v;
        Vcb[off] = (bf16)v;
      }
}

// ---------------- Kernel 2: LayerNorm(256) + Wk projection -> Kb[b][e][p] ----------------
__global__ __launch_bounds__(256) void kproj_kernel(const float* __restrict__ Vc,
                                                    const float* __restrict__ gx,
                                                    const float* __restrict__ bx,
                                                    const float* __restrict__ Wk,
                                                    const float* __restrict__ bk,
                                                    bf16* __restrict__ Kb) {
  __shared__ float t[32 * 257];  // 32 pixels x 256 ch (padded), pixel-major
  const int b = blockIdx.y, p0 = blockIdx.x * 32, tid = threadIdx.x;
  const float* V = Vc + (size_t)b * DX * P;

  for (int idx = tid; idx < DX * 32; idx += 256) {
    const int o = idx >> 5, px = idx & 31;
    t[px * 257 + o] = V[(size_t)o * P + p0 + px];
  }
  __syncthreads();

  // LayerNorm: 8 threads per pixel, 32 channels each
  {
    const int px = tid >> 3, j = tid & 7;
    float s = 0.f, s2 = 0.f;
    for (int i = 0; i < 32; ++i) {
      const float v = t[px * 257 + j * 32 + i];
      s += v; s2 += v * v;
    }
#pragma unroll
    for (int m = 1; m < 8; m <<= 1) { s += __shfl_xor(s, m, 32); s2 += __shfl_xor(s2, m, 32); }
    const float mu = s * (1.0f / DX);
    const float rs = rsqrtf(s2 * (1.0f / DX) - mu * mu + 1e-5f);
    for (int i = 0; i < 32; ++i) {
      const int ch = j * 32 + i;
      const float v = t[px * 257 + ch];
      t[px * 257 + ch] = (v - mu) * rs * gx[ch] + bx[ch];
    }
  }
  __syncthreads();

  const int wave = tid >> 5, mt = wave & 3, nt = wave >> 2;
  v8f acc = {};
  for (int kt = 0; kt < 8; ++kt) {
    v16bf a  = load_a_f32(Wk + (size_t)mt * 16 * DX, DX, kt * 32);
    v16bf bb = load_b_nmajor_f32(t, 257, kt * 32, nt * 16);
    acc = WMMA_BF16(a, bb, acc);
  }
  const int lane = tid & 31, n = lane & 15, hi = lane >> 4;
#pragma unroll
  for (int r = 0; r < 8; ++r) {
    const int e = mt * 16 + r + 8 * hi;
    const int p = p0 + nt * 16 + n;
    Kb[((size_t)b * E + e) * P + p] = (bf16)(acc[r] + bk[e]);
  }
}

// ---------------- Kernel 3: pixel-shuffle gather + LN(128) + Wq -> Qb[b][h][w][e] ----------------
__global__ __launch_bounds__(256) void qproj_kernel(const float* __restrict__ x,
                                                    const float* __restrict__ gy,
                                                    const float* __restrict__ by,
                                                    const float* __restrict__ Wq,
                                                    const float* __restrict__ bq,
                                                    bf16* __restrict__ Qb) {
  __shared__ float t[32 * 129];  // 32 w-positions x 128 ch (padded)
  const int b = blockIdx.z, h = blockIdx.y, w0 = blockIdx.x * 32, tid = threadIdx.x;
  const float* X  = x + (size_t)b * CIN * P;
  const int hoff = (h & 1) * 2;
  const int hp   = h >> 1;

  for (int idx = tid; idx < 32 * DY; idx += 256) {
    const int pxl = idx & 31, c = idx >> 5;
    const int w  = w0 + pxl;
    const int cx = c * 4 + hoff + (w & 1);   // pixel-shuffle channel
    t[pxl * 129 + c] = X[(size_t)cx * P + hp * Ww + (w >> 1)];
  }
  __syncthreads();

  {
    const int pxl = tid >> 3, j = tid & 7;
    float s = 0.f, s2 = 0.f;
    for (int i = 0; i < 16; ++i) {
      const float v = t[pxl * 129 + j * 16 + i];
      s += v; s2 += v * v;
    }
#pragma unroll
    for (int m = 1; m < 8; m <<= 1) { s += __shfl_xor(s, m, 32); s2 += __shfl_xor(s2, m, 32); }
    const float mu = s * (1.0f / DY);
    const float rs = rsqrtf(s2 * (1.0f / DY) - mu * mu + 1e-5f);
    for (int i = 0; i < 16; ++i) {
      const int ch = j * 16 + i;
      const float v = t[pxl * 129 + ch];
      t[pxl * 129 + ch] = (v - mu) * rs * gy[ch] + by[ch];
    }
  }
  __syncthreads();

  const int wave = tid >> 5, mt = wave & 3, nt = wave >> 2;
  v8f acc = {};
  for (int kt = 0; kt < 4; ++kt) {
    v16bf a  = load_a_f32(Wq + (size_t)mt * 16 * DY, DY, kt * 32);
    v16bf bb = load_b_nmajor_f32(t, 129, kt * 32, nt * 16);
    acc = WMMA_BF16(a, bb, acc);
  }
  const int lane = tid & 31, n = lane & 15, hi = lane >> 4;
  bf16* qbase = Qb + (((size_t)b * HU + h) * WU) * E;
#pragma unroll
  for (int r = 0; r < 8; ++r) {
    const int e = mt * 16 + r + 8 * hi;
    const int w = w0 + nt * 16 + n;
    qbase[(size_t)w * E + e] = (bf16)(acc[r] + bq[e]);
  }
}

// ---------------- Kernel 4: attention per (b,h): softmax(Q Kc^T/sqrt2) @ Vc ----------------
__global__ __launch_bounds__(256) void attn_kernel(const bf16* __restrict__ Qb,
                                                   const bf16* __restrict__ Kb,
                                                   const bf16* __restrict__ Vcb,
                                                   float* __restrict__ out) {
  __shared__ bf16 ksm[64 * 72];      // [u][e]  (n-major for B frags)
  __shared__ bf16 vs[256 * 64];      // [o][u]  (n-major for B frags)
  __shared__ bf16 ps[8 * 16 * 64];   // per-wave probs [16 rows][64 u]
  const int b = blockIdx.y, h = blockIdx.x, tid = threadIdx.x;
  const int pbase = (h >> 1) * Ww;   // unique-column base (upsample dedup)

  for (int idx = tid; idx < 64 * 64; idx += 256) {
    const int e = idx >> 6, u = idx & 63;
    ksm[u * 72 + e] = Kb[((size_t)b * E + e) * P + pbase + u];
  }
  for (int idx = tid; idx < 256 * 64; idx += 256) {
    const int o = idx >> 6, u = idx & 63;
    vs[o * 64 + u] = Vcb[((size_t)b * DX + o) * P + pbase + u];
  }
  __syncthreads();

  const int wave = tid >> 5, lane = tid & 31, n = lane & 15, hi = lane >> 4;
  const bf16* Qrow = Qb + (((size_t)b * HU + h) * WU + wave * 16) * E;

  // S = Q @ Kc^T  (128x64, each wave owns a 16-row strip)
  v8f sacc[4] = {};
#pragma unroll
  for (int kt = 0; kt < 2; ++kt) {
    v16bf a = load_a_bf(Qrow, E, kt * 32);
#pragma unroll
    for (int nt = 0; nt < 4; ++nt) {
      v16bf bb = load_b_nmajor_bf(ksm, 72, kt * 32, nt * 16);
      sacc[nt] = WMMA_BF16(a, bb, sacc[nt]);
    }
  }

  // Row softmax over 64 compressed logits (duplication cancels -> plain softmax)
  const float scale = 0.70710678f;  // 1/sqrt(head_size=2)
  bf16* pw = ps + wave * 16 * 64;
#pragma unroll
  for (int r = 0; r < 8; ++r) {
    float mx = -1e30f;
#pragma unroll
    for (int nt = 0; nt < 4; ++nt) mx = fmaxf(mx, sacc[nt][r] * scale);
#pragma unroll
    for (int m = 1; m < 16; m <<= 1) mx = fmaxf(mx, __shfl_xor(mx, m, 32));
    float e4[4]; float sum = 0.f;
#pragma unroll
    for (int nt = 0; nt < 4; ++nt) { e4[nt] = __expf(sacc[nt][r] * scale - mx); sum += e4[nt]; }
#pragma unroll
    for (int m = 1; m < 16; m <<= 1) sum += __shfl_xor(sum, m, 32);
    const float inv = 1.0f / sum;
    const int row = r + 8 * hi;
#pragma unroll
    for (int nt = 0; nt < 4; ++nt) pw[row * 64 + nt * 16 + n] = (bf16)(e4[nt] * inv);
  }
  __syncthreads();

  // O = P @ Vc  (16 x 256 per wave), store transposed output [b][o][w][h]
  v16bf pa0 = load_a_bf(pw, 64, 0);
  v16bf pa1 = load_a_bf(pw, 64, 32);
  float* outb = out + (size_t)b * DX * WU * HU;
#pragma unroll
  for (int ot = 0; ot < 16; ++ot) {
    v8f acc = {};
    v16bf b0 = load_b_nmajor_bf(vs, 64, 0, ot * 16);
    v16bf b1 = load_b_nmajor_bf(vs, 64, 32, ot * 16);
    acc = WMMA_BF16(pa0, b0, acc);
    acc = WMMA_BF16(pa1, b1, acc);
#pragma unroll
    for (int r = 0; r < 8; ++r) {
      const int o = ot * 16 + n;
      const int w = wave * 16 + r + 8 * hi;
      outb[((size_t)o * WU + w) * HU + h] = acc[r];
    }
  }
}

// ---------------- Host launcher ----------------
extern "C" void kernel_launch(void* const* d_in, const int* in_sizes, int n_in,
                              void* d_out, int out_size, void* d_ws, size_t ws_size,
                              hipStream_t stream) {
  (void)in_sizes; (void)n_in; (void)out_size; (void)ws_size;
  const float* x  = (const float*)d_in[0];
  const float* gy = (const float*)d_in[1];
  const float* by = (const float*)d_in[2];
  const float* gx = (const float*)d_in[3];
  const float* bx = (const float*)d_in[4];
  const float* Wq = (const float*)d_in[5];
  const float* bq = (const float*)d_in[6];
  const float* Wk = (const float*)d_in[7];
  const float* bk = (const float*)d_in[8];
  const float* Wc = (const float*)d_in[9];
  const float* bc = (const float*)d_in[10];
  float* out = (float*)d_out;

  char* ws = (char*)d_ws;
  float* Vc  = (float*)ws;                        // [8][256][4096] f32 : 32 MB
  bf16*  Vcb = (bf16*)(ws + 33554432);            // [8][256][4096] bf16: 16 MB
  bf16*  Kb  = (bf16*)(ws + 50331648);            // [8][64][4096]  bf16:  4 MB
  bf16*  Qb  = (bf16*)(ws + 54525952);            // [8][128][128][64] bf16: 16 MB

  conv1x1_kernel<<<dim3(64, Bn), 256, 0, stream>>>(x, Wc, bc, Vc, Vcb);
  kproj_kernel<<<dim3(128, Bn), 256, 0, stream>>>(Vc, gx, bx, Wk, bk, Kb);
  qproj_kernel<<<dim3(4, HU, Bn), 256, 0, stream>>>(x, gy, by, Wq, bq, Qb);
  attn_kernel<<<dim3(HU, Bn), 256, 0, stream>>>(Qb, Kb, Vcb, out);
}
